// F8Linear_51041391346217
// MI455X (gfx1250) — matrix operand order repaired and checked by reading, not compiled, not run on hardware
//
#include <hip/hip_runtime.h>
#include <stdint.h>

typedef __bf16 bf16_t;
typedef __attribute__((ext_vector_type(16))) __bf16 v16bf;
typedef __attribute__((ext_vector_type(8)))  __bf16 v8bf;
typedef __attribute__((ext_vector_type(4)))  __bf16 v4bf;
typedef __attribute__((ext_vector_type(8)))  float  v8f;
typedef __attribute__((ext_vector_type(4)))  float  v4f;
typedef __attribute__((ext_vector_type(4)))  int    v4i;

// async-copy builtin wants int4* pointers (per hipcc diagnostic); make AS-qualified ones
typedef __attribute__((address_space(1))) v4i* gptr_v4i;   // global
typedef __attribute__((address_space(3))) v4i* lptr_v4i;   // LDS

#define IN_F  4096
#define OUT_F 14336
#define BM 128
#define BN 128
#define BK 32
#define LDK 40              // padded LDS row stride (80B -> conflict-free b128 frag loads)
#define NK (IN_F / BK)      // 128 K-stages

#if __has_builtin(__builtin_amdgcn_global_load_async_to_lds_b128)
#define ASYNC_A 1
#else
#define ASYNC_A 0
#endif

#if __has_builtin(__builtin_amdgcn_s_wait_asynccnt)
#define WAIT_ASYNC() __builtin_amdgcn_s_wait_asynccnt(0)
#else
#define WAIT_ASYNC() asm volatile("s_wait_asynccnt 0" ::: "memory")
#endif

union FragBF { v16bf v; v8bf h[2]; };

__launch_bounds__(256)
__global__ void f8linear_wmma_bf16(const bf16_t* __restrict__ X,
                                   const float*  __restrict__ Wq,
                                   const float*  __restrict__ wscale,
                                   const float*  __restrict__ bias,
                                   bf16_t* __restrict__ Y)
{
    __shared__ __attribute__((aligned(16))) bf16_t As[2][BM * LDK];
    __shared__ __attribute__((aligned(16))) bf16_t Bs[2][BN * LDK];

    const int tid   = threadIdx.x;
    const int lane  = tid & 31;
    const int lrow  = lane & 15;
    const int lhalf = lane >> 4;        // wave32 lane half
    const int wave  = tid >> 5;         // 8 waves / block
    const int waveM = wave & 1;         // 2 waves along M (64 rows each)
    const int waveN = wave >> 1;        // 4 waves along N (32 cols each)

    // blockIdx.x walks M (fast): 8 consecutive blocks share one W tile -> W read once from HBM
    const int mBlk = blockIdx.x;
    const int nBlk = blockIdx.y;

    const float scale = wscale[0];      // folded into epilogue FMA (bf16(w_f8) is exact)

    const bf16_t* Ab = X  + (size_t)(mBlk * BM) * IN_F;
    const float*  Bb = Wq + (size_t)(nBlk * BN) * IN_F;

    // A tile: 128 rows x 32 bf16 = 512 x 16B chunks; thread owns {tid, tid+256}
    const int aR0 = tid >> 2;
    const int aR1 = (tid + 256) >> 2;
    const int aC  = (tid & 3) * 8;
    // B tile: 128 rows x 32 f32 = 1024 x 16B chunks; thread owns {tid + 256*j}
    int bR[4], bC[4];
#pragma unroll
    for (int j = 0; j < 4; ++j) { int c = tid + 256 * j; bR[j] = c >> 3; bC[j] = (c & 7) * 4; }

#if !ASYNC_A
    v4i aReg0, aReg1;
#endif
    v4f bReg[4];

    v8f acc[4][2];
#pragma unroll
    for (int mi = 0; mi < 4; ++mi)
#pragma unroll
        for (int ni = 0; ni < 2; ++ni)
#pragma unroll
            for (int e = 0; e < 8; ++e) acc[mi][ni][e] = 0.0f;

    auto loadB = [&](int kt) {
        const float* bp = Bb + (size_t)kt * BK;
#pragma unroll
        for (int j = 0; j < 4; ++j)
            bReg[j] = *(const v4f*)(bp + (size_t)bR[j] * IN_F + bC[j]);
    };
    auto storeB = [&](int buf) {
#pragma unroll
        for (int j = 0; j < 4; ++j) {
            v4bf w;
#pragma unroll
            for (int e = 0; e < 4; ++e) w[e] = (__bf16)bReg[j][e];   // exact: fp8 subset of bf16
            *(v4bf*)&Bs[buf][bR[j] * LDK + bC[j]] = w;
        }
    };
#if ASYNC_A
    auto copyA = [&](int buf, int kt) {   // async global -> LDS, ASYNCcnt-tracked
        const bf16_t* ap = Ab + (size_t)kt * BK;
        __builtin_amdgcn_global_load_async_to_lds_b128(
            (gptr_v4i)(ap + (size_t)aR0 * IN_F + aC),
            (lptr_v4i)&As[buf][aR0 * LDK + aC], 0, 0);
        __builtin_amdgcn_global_load_async_to_lds_b128(
            (gptr_v4i)(ap + (size_t)aR1 * IN_F + aC),
            (lptr_v4i)&As[buf][aR1 * LDK + aC], 0, 0);
    };
#else
    auto loadA = [&](int kt) {
        const bf16_t* ap = Ab + (size_t)kt * BK;
        aReg0 = *(const v4i*)(ap + (size_t)aR0 * IN_F + aC);
        aReg1 = *(const v4i*)(ap + (size_t)aR1 * IN_F + aC);
    };
    auto storeA = [&](int buf) {
        *(v4i*)&As[buf][aR0 * LDK + aC] = aReg0;
        *(v4i*)&As[buf][aR1 * LDK + aC] = aReg1;
    };
#endif

    // ---- prologue: stage tile 0 into buffer 0
#if ASYNC_A
    copyA(0, 0);
#else
    loadA(0); storeA(0);
#endif
    loadB(0); storeB(0);
#if ASYNC_A
    WAIT_ASYNC();
#endif
    __syncthreads();

    for (int kt = 0; kt < NK; ++kt) {
        const int cur = kt & 1;
        const int nxt = cur ^ 1;

        if (kt + 1 < NK) {
#if ASYNC_A
            copyA(nxt, kt + 1);          // writes LDS[nxt]; readers finished before last barrier
#else
            loadA(kt + 1);
#endif
            loadB(kt + 1);
        }
        if (kt + 2 < NK) {               // global_prefetch two tiles ahead (keep near: RT)
            __builtin_prefetch(Ab + (size_t)(kt + 2) * BK + (size_t)aR0 * IN_F, 0, 3);
            __builtin_prefetch(Bb + (size_t)(kt + 2) * BK + (size_t)bR[0] * IN_F, 0, 3);
        }

        // ---- WMMA over LDS[cur]
        FragBF a[4], b[2];
#pragma unroll
        for (int mi = 0; mi < 4; ++mi) {
            const int r = waveM * 64 + mi * 16 + lrow;
            // 16-bit A layout: lanes 0-15 -> K {0..7,16..23}; lanes 16-31 -> K {8..15,24..31}
            a[mi].h[0] = *(const v8bf*)&As[cur][r * LDK + lhalf * 8];
            a[mi].h[1] = *(const v8bf*)&As[cur][r * LDK + lhalf * 8 + 16];
        }
#pragma unroll
        for (int ni = 0; ni < 2; ++ni) {
            const int r = waveN * 32 + ni * 16 + lrow;
            // 16-bit B layout: lanes 0-15 -> K {0..7,16..23}; lanes 16-31 -> K {8..15,24..31}
            b[ni].h[0] = *(const v8bf*)&Bs[cur][r * LDK + lhalf * 8];
            b[ni].h[1] = *(const v8bf*)&Bs[cur][r * LDK + lhalf * 8 + 16];
        }
#pragma unroll
        for (int mi = 0; mi < 4; ++mi)
#pragma unroll
            for (int ni = 0; ni < 2; ++ni)
                acc[mi][ni] = __builtin_amdgcn_wmma_f32_16x16x32_bf16(
                    false, a[mi].v, false, b[ni].v, (short)0, acc[mi][ni], false, false);

        // publish next tile
        if (kt + 1 < NK) {
#if !ASYNC_A
            storeA(nxt);
#endif
            storeB(nxt);
        }
#if ASYNC_A
        WAIT_ASYNC();                    // this wave's async A copies landed in LDS
#endif
        __syncthreads();
    }

    // ---- epilogue: y = acc*scale + bias, round to bf16
    // C layout: VGPR r, lane half -> M = r + 8*lhalf, N = lane%16
    const int nBase = nBlk * BN + waveN * 32;
    const int mBase = mBlk * BM + waveM * 64;
#pragma unroll
    for (int ni = 0; ni < 2; ++ni) {
        const int col = nBase + ni * 16 + lrow;
        const float bv = bias[col];
#pragma unroll
        for (int mi = 0; mi < 4; ++mi) {
            const int row0 = mBase + mi * 16 + lhalf * 8;
#pragma unroll
            for (int r = 0; r < 8; ++r)
                Y[(size_t)(row0 + r) * OUT_F + col] = (__bf16)(acc[mi][ni][r] * scale + bv);
        }
    }
}

extern "C" void kernel_launch(void* const* d_in, const int* in_sizes, int n_in,
                              void* d_out, int out_size, void* d_ws, size_t ws_size,
                              hipStream_t stream)
{
    const bf16_t* x  = (const bf16_t*)d_in[0];   // [2,512,4096] bf16
    const float*  wq = (const float*) d_in[1];   // [14336,4096] f32 (fp8-representable)
    const float*  ws = (const float*) d_in[2];   // scalar scale
    const float*  bs = (const float*) d_in[3];   // [14336] f32
    bf16_t*       y  = (bf16_t*)d_out;           // [2,512,14336] bf16

    const int M = in_sizes[0] / IN_F;            // 1024
    dim3 grid(M / BM, OUT_F / BN);               // 8 x 112: M fastest -> W tile reused from L2
    f8linear_wmma_bf16<<<grid, dim3(256), 0, stream>>>(x, wq, ws, bs, y);
}
